// FastplaneModule_28312424415680
// MI455X (gfx1250) — matrix operand order repaired and checked by reading, not compiled, not run on hardware
//
#include <hip/hip_runtime.h>
#include <hip/hip_bf16.h>

typedef __attribute__((ext_vector_type(16))) __bf16 v16bf;
typedef __attribute__((ext_vector_type(8)))  __bf16 v8bf;
typedef __attribute__((ext_vector_type(8)))  float  v8f;

#define C_DIM 64
#define HW    256
#define S_SAMP 64
#define KRD   16
#define NRAYS 16384    // B*R = 2*8192
#define HSTRIDE 72     // bf16 row stride: 144B = 9*16B -> aligned, conflict-free b128
#define GAIN  1.0f

static __device__ __forceinline__ __bf16 f2bf(float f) {
  unsigned u = __builtin_bit_cast(unsigned, f);
  unsigned r = u + 0x7FFFu + ((u >> 16) & 1u);   // round-to-nearest-even
  unsigned short h = (unsigned short)(r >> 16);
  return __builtin_bit_cast(__bf16, h);
}

static __device__ __forceinline__ float bf2f(__bf16 b) {
  unsigned short h = __builtin_bit_cast(unsigned short, b);
  unsigned u = ((unsigned)h) << 16;
  return __builtin_bit_cast(float, u);
}

// fast stable softplus: max(x,0) + log(1 + exp(-|x|)); v_exp_f32/v_log_f32 HW approx
static __device__ __forceinline__ float softplus_f(float x) {
  return fmaxf(x, 0.f) + __logf(1.f + __expf(-fabsf(x)));
}

// one bilinear tap footprint shared by all planes sampled at the same (u,v)
struct Tap { int off; float w00, w01, w10, w11; };

static __device__ __forceinline__ Tap make_tap(float u, float v) {
  u = fminf(fmaxf(u, -1.f), 1.f);
  v = fminf(fmaxf(v, -1.f), 1.f);
  const float x = (u + 1.f) * 0.5f * (float)(HW - 1);
  const float y = (v + 1.f) * 0.5f * (float)(HW - 1);
  const int x0 = min(max((int)floorf(x), 0), HW - 2);
  const int y0 = min(max((int)floorf(y), 0), HW - 2);
  const float wx = x - (float)x0, wy = y - (float)y0;
  Tap t;
  t.off = y0 * HW + x0;
  t.w11 = wx * wy;
  t.w01 = wx - t.w11;              // wx*(1-wy)
  t.w10 = wy - t.w11;              // (1-wx)*wy
  t.w00 = 1.f - wx - wy + t.w11;   // (1-wx)*(1-wy)
  return t;
}

static __device__ __forceinline__ float tap_eval(const float* __restrict__ plane,
                                                 int c, const Tap& t) {
  const float* p = plane + ((size_t)c << 16) + t.off;  // c*HW*HW
  return p[0] * t.w00 + p[1] * t.w01 + p[HW] * t.w10 + p[HW + 1] * t.w11;
}

static __device__ __forceinline__ v8bf lds_v8(const __bf16* p) {
  return *reinterpret_cast<const v8bf*>(p);
}

static __device__ __forceinline__ v16bf cat16(v8bf lo, v8bf hi) {
  return __builtin_shufflevector(lo, hi, 0, 1, 2, 3, 4, 5, 6, 7,
                                         8, 9, 10, 11, 12, 13, 14, 15);
}

static __device__ __forceinline__ v8f wmma_bf16(v16bf a, v16bf b, v8f c) {
  // D = A(16x32 bf16) * B(32x16 bf16) + C(16x16 f32)
  return __builtin_amdgcn_wmma_f32_16x16x32_bf16(false, a, false, b, (short)0, c,
                                                 false, false);
}

__global__ __launch_bounds__(128) void fastplane_kernel(
    const float* __restrict__ rays, const float* __restrict__ centers,
    const float* __restrict__ enc,  const float* __restrict__ nearp,
    const float* __restrict__ farp,
    const float* __restrict__ xy,  const float* __restrict__ yz,  const float* __restrict__ zx,
    const float* __restrict__ xyc, const float* __restrict__ yzc, const float* __restrict__ zxc,
    const float* __restrict__ mlpw, const float* __restrict__ mlpb,
    const float* __restrict__ wop,  const float* __restrict__ bop,
    const float* __restrict__ wc,   const float* __restrict__ bc,
    const float* __restrict__ bg,   float* __restrict__ out)
{
  __shared__ __align__(16) __bf16 h_bf[S_SAMP * HSTRIDE];      // hidden feats [64][72]
  __shared__ __align__(16) __bf16 fc_bf[S_SAMP * HSTRIDE];     // color feats + enc
  __shared__ __align__(16) __bf16 wT_lds[2 * C_DIM * HSTRIDE]; // W^T[n][k] per layer
  __shared__ __align__(16) __bf16 wcT_lds[KRD * HSTRIDE];      // Wc^T[n][k]
  __shared__ float wop_lds[C_DIM];
  __shared__ float bias_lds[2 * C_DIM];
  __shared__ float bc_lds[KRD];
  __shared__ float bg_lds[KRD];
  __shared__ float sd_lds[S_SAMP];
  __shared__ float csd_lds[S_SAMP];
  __shared__ float w_lds[S_SAMP];
  __shared__ float feat_lds[KRD];
  __shared__ float mask_lds;
  __shared__ float raylen_lds;

  const int tid   = threadIdx.x;
  const int lane  = tid & 31;
  const int wave  = tid >> 5;       // 0..3; each wave owns 16 samples of the ray
  const int sbase = wave * 16;
  const int ridx  = blockIdx.x;     // flat over B*R
  const int bat   = ridx >> 13;     // /8192

  // ---- stage weights transposed into LDS (bf16) ----
  for (int idx = tid; idx < 2 * C_DIM * C_DIM; idx += 128) {
    const int l = idx >> 12;
    const int rem = idx & 4095;
    const int k = rem >> 6;
    const int n = rem & 63;
    wT_lds[(l * C_DIM + n) * HSTRIDE + k] = f2bf(mlpw[idx]);
  }
  for (int idx = tid; idx < C_DIM * KRD; idx += 128) {
    const int k = idx >> 4;
    const int n = idx & 15;
    wcT_lds[n * HSTRIDE + k] = f2bf(wc[idx]);
  }
  if (tid < C_DIM) {
    wop_lds[tid]          = wop[tid];
    bias_lds[tid]         = mlpb[tid];
    bias_lds[C_DIM + tid] = mlpb[C_DIM + tid];
  }
  if (tid < KRD) { bc_lds[tid] = bc[tid]; bg_lds[tid] = bg[tid]; feat_lds[tid] = 0.f; }
  if (tid == 0)  { raylen_lds = 0.f; }

  // ---- per-ray scalars (uniform across block) ----
  const float nearv = nearp[ridx];
  const float farv  = farp[ridx];
  const float cx = centers[ridx * 3 + 0], cy = centers[ridx * 3 + 1], cz = centers[ridx * 3 + 2];
  const float dx = rays[ridx * 3 + 0],    dy = rays[ridx * 3 + 1],    dz = rays[ridx * 3 + 2];
  const float delta = (farv - nearv) * (1.f / (float)S_SAMP);
  const float bopv  = bop[0];

  const size_t pbase = (size_t)bat * C_DIM * HW * HW;
  const float* xy_b  = xy  + pbase;
  const float* yz_b  = yz  + pbase;
  const float* zx_b  = zx  + pbase;
  const float* xyc_b = xyc + pbase;
  const float* yzc_b = yzc + pbase;
  const float* zxc_b = zxc + pbase;

  // per-lane ray-encoding channels (folded into color features)
  const float enc0 = enc[(size_t)ridx * C_DIM + lane];
  const float enc1 = enc[(size_t)ridx * C_DIM + lane + 32];

  // ---- phase 1: triplane bilinear sampling ----
  // taps for sample ms are computed once (3 orientations), shared by 6 planes and
  // both channel halves; taps for ms+1 drive prefetch and are reused next iter.
  Tap txy, tyz, tzx;
  {
    const float tt = nearv + (farv - nearv) * (((float)sbase + 0.5f) * (1.f / (float)S_SAMP));
    const float px = cx + tt * dx, py = cy + tt * dy, pz = cz + tt * dz;
    txy = make_tap(px, py); tyz = make_tap(py, pz); tzx = make_tap(pz, px);
  }
  for (int ms = 0; ms < 16; ++ms) {
    const int s = sbase + ms;
    #pragma unroll
    for (int ch = 0; ch < 2; ++ch) {
      const int c = lane + 32 * ch;
      const float hv = tap_eval(xy_b, c, txy)
                     + tap_eval(yz_b, c, tyz)
                     + tap_eval(zx_b, c, tzx);
      h_bf[s * HSTRIDE + c] = f2bf(hv);
      const float fv = tap_eval(xyc_b, c, txy)
                     + tap_eval(yzc_b, c, tyz)
                     + tap_eval(zxc_b, c, tzx)
                     + (ch ? enc1 : enc0);
      fc_bf[s * HSTRIDE + c] = f2bf(fv);
    }
    if (ms < 15) {
      const float t2 = nearv + (farv - nearv) * (((float)s + 1.5f) * (1.f / (float)S_SAMP));
      const float qx = cx + t2 * dx, qy = cy + t2 * dy, qz = cz + t2 * dz;
      txy = make_tap(qx, qy); tyz = make_tap(qy, qz); tzx = make_tap(qz, qx);
      const size_t cb = (size_t)lane << 16;
      __builtin_prefetch(xy_b  + cb + txy.off, 0, 1);
      __builtin_prefetch(yz_b  + cb + tyz.off, 0, 1);
      __builtin_prefetch(zx_b  + cb + tzx.off, 0, 1);
      __builtin_prefetch(xyc_b + cb + txy.off, 0, 1);
      __builtin_prefetch(yzc_b + cb + tyz.off, 0, 1);
      __builtin_prefetch(zxc_b + cb + tzx.off, 0, 1);
    }
  }

  __syncthreads();  // weights staged; features in LDS

  // WMMA fragment indexing (ISA 7.12.2, wave32)
  const int mrow = lane & 15;            // A: row per lane
  const int kb_a = (lane < 16) ? 0 : 8;  // A: K base by lane half
  const int kb_b = (lane < 16) ? 0 : 8;  // B: K base by lane half
  const int ncol = lane & 15;            // B/D: column per lane
  const int rh   = lane >> 4;            // D: row-half per lane

  const __bf16* hrow = &h_bf[(sbase + mrow) * HSTRIDE];

  // ---- phase 2: 2-layer MLP via WMMA; sigma folded into layer-2 epilogue ----
  #pragma unroll
  for (int l = 0; l < 2; ++l) {
    const v16bf a0 = cat16(lds_v8(hrow + kb_a),      lds_v8(hrow + kb_a + 16));
    const v16bf a1 = cat16(lds_v8(hrow + kb_a + 32), lds_v8(hrow + kb_a + 48));
    float sigp[8];
    #pragma unroll
    for (int v = 0; v < 8; ++v) sigp[v] = 0.f;
    #pragma unroll
    for (int t = 0; t < 4; ++t) {
      const __bf16* wrow = &wT_lds[(l * C_DIM + t * 16 + ncol) * HSTRIDE];
      const v16bf b0 = cat16(lds_v8(wrow + kb_b),      lds_v8(wrow + kb_b + 16));
      const v16bf b1 = cat16(lds_v8(wrow + kb_b + 32), lds_v8(wrow + kb_b + 48));
      v8f acc = {0.f, 0.f, 0.f, 0.f, 0.f, 0.f, 0.f, 0.f};
      acc = wmma_bf16(a0, b0, acc);
      acc = wmma_bf16(a1, b1, acc);
      const float bias = bias_lds[l * C_DIM + t * 16 + ncol];
      const float wopv = wop_lds[t * 16 + ncol];
      #pragma unroll
      for (int v = 0; v < 8; ++v) {
        const float r = softplus_f(GAIN * (acc[v] + bias));
        h_bf[(sbase + v + 8 * rh) * HSTRIDE + t * 16 + ncol] = f2bf(r);
        if (l == 1) sigp[v] += r * wopv;
      }
    }
    if (l == 1) {
      // reduce sigma partials across the 16 columns of each lane-half
      #pragma unroll
      for (int v = 0; v < 8; ++v) {
        float pv = sigp[v];
        pv += __shfl_xor(pv, 1, 16);
        pv += __shfl_xor(pv, 2, 16);
        pv += __shfl_xor(pv, 4, 16);
        pv += __shfl_xor(pv, 8, 16);
        const float sig = softplus_f(GAIN * (pv + bopv));
        if (ncol == 0) sd_lds[sbase + v + 8 * rh] = sig * delta;
      }
    }
  }

  // ---- color projection: (h2 + fc + enc) @ Wc  (two WMMAs, N=16) ----
  const __bf16* frow = &fc_bf[(sbase + mrow) * HSTRIDE];
  const v16bf hA0 = cat16(lds_v8(hrow + kb_a),      lds_v8(hrow + kb_a + 16));
  const v16bf hA1 = cat16(lds_v8(hrow + kb_a + 32), lds_v8(hrow + kb_a + 48));
  const v16bf fA0 = cat16(lds_v8(frow + kb_a),      lds_v8(frow + kb_a + 16));
  const v16bf fA1 = cat16(lds_v8(frow + kb_a + 32), lds_v8(frow + kb_a + 48));
  v16bf ac0, ac1;
  #pragma unroll
  for (int i = 0; i < 16; ++i) {
    ac0[i] = f2bf(bf2f(hA0[i]) + bf2f(fA0[i]));
    ac1[i] = f2bf(bf2f(hA1[i]) + bf2f(fA1[i]));
  }
  const __bf16* wrc = &wcT_lds[ncol * HSTRIDE];
  const v16bf cb0 = cat16(lds_v8(wrc + kb_b),      lds_v8(wrc + kb_b + 16));
  const v16bf cb1 = cat16(lds_v8(wrc + kb_b + 32), lds_v8(wrc + kb_b + 48));
  v8f cacc = {0.f, 0.f, 0.f, 0.f, 0.f, 0.f, 0.f, 0.f};
  cacc = wmma_bf16(ac0, cb0, cacc);
  cacc = wmma_bf16(ac1, cb1, cacc);
  float colreg[8];
  #pragma unroll
  for (int v = 0; v < 8; ++v) colreg[v] = cacc[v] + bc_lds[ncol];

  // ---- transmittance scan over the full ray ----
  __syncthreads();
  if (tid == 0) {
    float cs = 0.f;
    for (int s = 0; s < S_SAMP; ++s) { cs += sd_lds[s]; csd_lds[s] = cs; }
    mask_lds = 1.f - __expf(-cs);
  }
  __syncthreads();

  if (lane < 16) {
    const int s = sbase + lane;
    const float sd  = sd_lds[s];
    const float wgt = __expf(sd - csd_lds[s]) * (1.f - __expf(-sd));
    w_lds[s] = wgt;
    const float tt = nearv + (farv - nearv) * (((float)s + 0.5f) * (1.f / (float)S_SAMP));
    atomicAdd(&raylen_lds, wgt * tt);
  }

  // feat[k] = sum_s w[s] * color[s][k]  (lane holds column ncol, 8 rows)
  float fsum = 0.f;
  #pragma unroll
  for (int v = 0; v < 8; ++v) fsum += w_lds[sbase + v + 8 * rh] * colreg[v];
  atomicAdd(&feat_lds[ncol], fsum);

  __syncthreads();

  // ---- outputs: feat [NRAYS,16] ++ mask [NRAYS] ++ ray_len [NRAYS] ----
  if (tid < KRD) {
    const float f = feat_lds[tid] + (1.f - mask_lds) * bg_lds[tid];
    out[(size_t)ridx * KRD + tid] = f;
  }
  if (tid == 0) {
    out[(size_t)NRAYS * KRD + ridx]         = mask_lds;
    out[(size_t)NRAYS * KRD + NRAYS + ridx] = raylen_lds;
  }
}

extern "C" void kernel_launch(void* const* d_in, const int* in_sizes, int n_in,
                              void* d_out, int out_size, void* d_ws, size_t ws_size,
                              hipStream_t stream) {
  (void)in_sizes; (void)n_in; (void)out_size; (void)d_ws; (void)ws_size;
  const float* rays    = (const float*)d_in[0];
  const float* centers = (const float*)d_in[1];
  const float* enc     = (const float*)d_in[2];
  const float* nearp   = (const float*)d_in[3];
  const float* farp    = (const float*)d_in[4];
  const float* xy      = (const float*)d_in[5];
  const float* yz      = (const float*)d_in[6];
  const float* zx      = (const float*)d_in[7];
  const float* xyc     = (const float*)d_in[8];
  const float* yzc     = (const float*)d_in[9];
  const float* zxc     = (const float*)d_in[10];
  const float* mlpw    = (const float*)d_in[11];
  const float* mlpb    = (const float*)d_in[12];
  const float* wop     = (const float*)d_in[13];
  const float* bop     = (const float*)d_in[14];
  const float* wc      = (const float*)d_in[15];
  const float* bc      = (const float*)d_in[16];
  const float* bg      = (const float*)d_in[17];
  float* out = (float*)d_out;

  fastplane_kernel<<<NRAYS, 128, 0, stream>>>(
      rays, centers, enc, nearp, farp, xy, yz, zx, xyc, yzc, zxc,
      mlpw, mlpb, wop, bop, wc, bc, bg, out);
}
